// QuantumADMETModel_67800353734920
// MI455X (gfx1250) — matrix operand order repaired and checked by reading, not compiled, not run on hardware
//
#include <hip/hip_runtime.h>
#include <cstdint>
#include <cstddef>

// ---------------------------------------------------------------------------
// QuantumADMET GNN forward for MI455X (gfx1250, wave32, WMMA).
//
// Main GEMMs ([32768,256]x[256,256] etc.) run on v_wmma_f32_16x16x32_f16:
// f16 inputs, f32 accumulation. W K-slices are double-buffered in LDS with
// split load/store staging so global latency hides behind the WMMA chain.
// Edge attention uses a 3-pass segment softmax with global float atomics.
// LN/GELU/pool/towers are VALU kernels.
// ---------------------------------------------------------------------------

#define NN     32768      // nodes
#define NE     262144     // edges
#define NG     1024       // graphs
#define HID    256
#define NHEAD  8
#define DHEAD  32
#define NLAYER 5
#define EGAT   (NE + NN)  // edges + self loops for GATv2

typedef __attribute__((ext_vector_type(16))) _Float16     v16h;
typedef __attribute__((ext_vector_type(4)))  _Float16     v4h;
typedef __attribute__((ext_vector_type(8)))  float        v8f;
typedef __attribute__((ext_vector_type(4)))  unsigned int v4u;

__device__ __forceinline__ float geluf(float x) {
  return 0.5f * x * (1.0f + erff(x * 0.70710678118654752f));
}
// monotonic float<->int encoding for atomicMax on floats
__device__ __forceinline__ int   fenc(float f) { int i = __float_as_int(f); return i >= 0 ? i : (i ^ 0x7fffffff); }
__device__ __forceinline__ float fdec(int i)   { return __int_as_float(i >= 0 ? i : (i ^ 0x7fffffff)); }

// ---------------------------------------------------------------------------
// Fill helpers
// ---------------------------------------------------------------------------
__global__ void k_fill_f(float* __restrict__ p, float v, int n) {
  int i = blockIdx.x * blockDim.x + threadIdx.x;
  if (i < n) p[i] = v;
}
__global__ void k_fill_i(int* __restrict__ p, int v, int n) {
  int i = blockIdx.x * blockDim.x + threadIdx.x;
  if (i < n) p[i] = v;
}
// Y[n][c] = bias[c] for all rows (init of attention output accumulator)
__global__ void k_rows_bias(float* __restrict__ Y, const float* __restrict__ bias, int rows) {
  int i = blockIdx.x * blockDim.x + threadIdx.x;
  if (i < rows * HID) Y[i] = bias[i & (HID - 1)];
}

// ---------------------------------------------------------------------------
// Input embed: h = GELU(LN(x @ W + b));  x:[NN,34], W:[34,256]
// One block per node, 256 threads.
// ---------------------------------------------------------------------------
__global__ __launch_bounds__(256) void k_embed(
    const float* __restrict__ x, const float* __restrict__ W,
    const float* __restrict__ b, const float* __restrict__ g,
    const float* __restrict__ be, float* __restrict__ h) {
  __shared__ float xs[34];
  __shared__ float red[256];
  const int n = blockIdx.x, t = threadIdx.x;
  if (t < 34) xs[t] = x[n * 34 + t];
  __syncthreads();
  float acc = b[t];
#pragma unroll 2
  for (int i = 0; i < 34; ++i) acc += xs[i] * W[i * HID + t];
  // LayerNorm over 256
  red[t] = acc; __syncthreads();
  for (int off = 128; off > 0; off >>= 1) { if (t < off) red[t] += red[t + off]; __syncthreads(); }
  float mu = red[0] * (1.0f / HID); __syncthreads();
  float d = acc - mu;
  red[t] = d * d; __syncthreads();
  for (int off = 128; off > 0; off >>= 1) { if (t < off) red[t] += red[t + off]; __syncthreads(); }
  float var = red[0] * (1.0f / HID);
  float y = d * rsqrtf(var + 1e-5f) * g[t] + be[t];
  h[(size_t)n * HID + t] = geluf(y);
}

// ---------------------------------------------------------------------------
// WMMA GEMM: Y[M,NW] = X[M,K] @ W[K,NW] + bias.  K % 32 == 0, NW % 256 == 0,
// M % 128 == 0.  grid = (M/128, NW/256), block = 256 (8 waves).
// Per wave: 16 rows x 256 cols = 16 accumulators of v8f.
// W K-slices double-buffered in LDS as f16 (padded stride 264); staging is
// split into a load phase (before compute, latency hidden by 16 WMMAs) and a
// convert/store phase (after compute). One barrier per K-step.
// ---------------------------------------------------------------------------
__global__ __launch_bounds__(256) void k_gemm(
    const float* __restrict__ X, const float* __restrict__ W,
    const float* __restrict__ bias, float* __restrict__ Y,
    int M, int K, int NW) {
  __shared__ _Float16 Bs[2][32][264];
  const int tid  = threadIdx.x;
  const int wave = tid >> 5;
  const int lane = tid & 31;
  const int half = lane >> 4;   // 0: lanes 0-15, 1: lanes 16-31
  const int l16  = lane & 15;
  const int colBase = blockIdx.y * 256;
  const int rowBase = blockIdx.x * 128 + wave * 16;

  v8f acc[16];
#pragma unroll
  for (int t = 0; t < 16; ++t)
#pragma unroll
    for (int r = 0; r < 8; ++r) acc[t][r] = 0.0f;

  // Staging registers: 8 float4 per thread = one 32x256 W slice per block.
  float4 wreg[8];
  auto load_slice = [&](int k0) {
#pragma unroll
    for (int i = 0; i < 8; ++i) {
      int idx = tid + i * 256;          // float4 index within slice (0..2047)
      int kk  = idx >> 6;               // 64 float4 per K-row
      int nn  = (idx & 63) * 4;
      wreg[i] = *(const float4*)&W[(size_t)(k0 + kk) * NW + colBase + nn];
    }
  };
  auto store_slice = [&](int buf) {
#pragma unroll
    for (int i = 0; i < 8; ++i) {
      int idx = tid + i * 256;
      int kk  = idx >> 6;
      int nn  = (idx & 63) * 4;
      v4h hv;
      hv[0] = (_Float16)wreg[i].x; hv[1] = (_Float16)wreg[i].y;
      hv[2] = (_Float16)wreg[i].z; hv[3] = (_Float16)wreg[i].w;
      *(v4h*)&Bs[buf][kk][nn] = hv;     // 8B-aligned ds_store_b64
    }
  };

  load_slice(0);
  store_slice(0);
  __syncthreads();

  const int nsteps = K >> 5;
  for (int s = 0; s < nsteps; ++s) {
    const int k0  = s << 5;
    const int cur = s & 1;

    // Issue global loads for the next W slice; latency hides behind compute.
    if (s + 1 < nsteps) load_slice(k0 + 32);

    // A fragment (16-bit A 16x32 layout): row = rowBase+l16,
    //   elems 0..7  -> K = k0 + 8*half + {0..7}
    //   elems 8..15 -> K = k0 + 16 + 8*half + {0..7}
    const float* xp = X + (size_t)(rowBase + l16) * K + k0 + half * 8;
    v16h a;
#pragma unroll
    for (int j = 0; j < 8; ++j) a[j] = (_Float16)xp[j];
#pragma unroll
    for (int j = 0; j < 8; ++j) a[8 + j] = (_Float16)xp[16 + j];

    // Rotating 2-deep B-fragment pipeline over the 16 N-tiles.
    union BF { v4u u[2]; v16h h; };
    BF bf[2];
    {
      const _Float16* bp = &Bs[cur][half * 8][l16];
#pragma unroll
      for (int j = 0; j < 8; ++j) { bf[0].h[j] = bp[j * 264]; bf[0].h[8 + j] = bp[(16 + j) * 264]; }
    }
#pragma unroll
    for (int t = 0; t < 16; ++t) {
      if (t < 15) {
        const _Float16* bp = &Bs[cur][half * 8][(t + 1) * 16 + l16];
#pragma unroll
        for (int j = 0; j < 8; ++j) { bf[(t + 1) & 1].h[j] = bp[j * 264]; bf[(t + 1) & 1].h[8 + j] = bp[(16 + j) * 264]; }
      }
      acc[t] = __builtin_amdgcn_wmma_f32_16x16x32_f16(
          false, a, false, bf[t & 1].h, (short)0, acc[t], false, false);
    }

    // Convert + store next slice (global loads completed during compute).
    if (s + 1 < nsteps) store_slice(cur ^ 1);
    __syncthreads();
  }

  // D layout: vgpr r, lanes 0-15 -> M=r, lanes 16-31 -> M=r+8, N = l16
  const int orow = rowBase + half * 8;
#pragma unroll
  for (int t = 0; t < 16; ++t) {
    int col = colBase + t * 16 + l16;
    float bv = bias[col];
#pragma unroll
    for (int r = 0; r < 8; ++r)
      Y[(size_t)(orow + r) * NW + col] = acc[t][r] + bv;
  }
}

// ---------------------------------------------------------------------------
// Attention pass A (GATv2): logit[e][h] = att[h] . leaky(xl[src]+xr[dst]);
// one wave per edge; lane: head = lane/4, sub = lane%4 covers 8 dims.
// Also atomic-max into per-(dst,head) running max (int-encoded).
// ---------------------------------------------------------------------------
__global__ __launch_bounds__(256) void k_attA_gat(
    const float* __restrict__ xl, const float* __restrict__ xr,
    const float* __restrict__ att, const int* __restrict__ ei,
    int E, int Etot, float* __restrict__ logits, int* __restrict__ mEnc) {
  int gw = (blockIdx.x * blockDim.x + threadIdx.x) >> 5;
  if (gw >= Etot) return;
  int lane = threadIdx.x & 31;
  int hh = lane >> 2, q = lane & 3;
  int s, d;
  if (gw < E) { s = ei[gw]; d = ei[E + gw]; } else { s = gw - E; d = s; }
  const float* pa = xl + (size_t)s * HID + hh * DHEAD + q * 8;
  const float* pb = xr + (size_t)d * HID + hh * DHEAD + q * 8;
  const float* pt = att + hh * DHEAD + q * 8;
  float p = 0.0f;
#pragma unroll
  for (int j = 0; j < 8; ++j) {
    float v = pa[j] + pb[j];
    v = v > 0.0f ? v : 0.2f * v;   // leaky_relu(0.2)
    p += v * pt[j];
  }
  p += __shfl_xor(p, 1, 32);
  p += __shfl_xor(p, 2, 32);
  if (q == 0) {
    logits[(size_t)gw * NHEAD + hh] = p;
    atomicMax(mEnc + (size_t)d * NHEAD + hh, fenc(p));
  }
}

// Attention pass A (TransformerConv): logit = q[dst].k[src] / sqrt(DH)
__global__ __launch_bounds__(256) void k_attA_trans(
    const float* __restrict__ Q, const float* __restrict__ Kt,
    const int* __restrict__ ei, int E,
    float* __restrict__ logits, int* __restrict__ mEnc) {
  int gw = (blockIdx.x * blockDim.x + threadIdx.x) >> 5;
  if (gw >= E) return;
  int lane = threadIdx.x & 31;
  int hh = lane >> 2, q = lane & 3;
  int s = ei[gw], d = ei[E + gw];
  const float* pq = Q  + (size_t)d * HID + hh * DHEAD + q * 8;
  const float* pk = Kt + (size_t)s * HID + hh * DHEAD + q * 8;
  float p = 0.0f;
#pragma unroll
  for (int j = 0; j < 8; ++j) p += pq[j] * pk[j];
  p += __shfl_xor(p, 1, 32);
  p += __shfl_xor(p, 2, 32);
  p *= 0.17677669529663687f;  // 1/sqrt(32)
  if (q == 0) {
    logits[(size_t)gw * NHEAD + hh] = p;
    atomicMax(mEnc + (size_t)d * NHEAD + hh, fenc(p));
  }
}

// Pass B: w = exp(logit - max[dst]); accumulate sums. One thread per (e,h).
__global__ void k_attB(float* __restrict__ logits, const int* __restrict__ mEnc,
                       float* __restrict__ ssum, const int* __restrict__ ei,
                       int E, int Etot) {
  int i = blockIdx.x * blockDim.x + threadIdx.x;
  if (i >= Etot * NHEAD) return;
  int e = i >> 3, hh = i & 7;
  int d = (e < E) ? ei[E + e] : (e - E);
  float w = expf(logits[i] - fdec(mEnc[(size_t)d * NHEAD + hh]));
  logits[i] = w;
  atomicAdd(ssum + (size_t)d * NHEAD + hh, w);
}

// Pass C: out[dst] += val[src] * (w / sum[dst]). One wave per edge.
__global__ __launch_bounds__(256) void k_attC(
    const float* __restrict__ val, const float* __restrict__ logits,
    const float* __restrict__ ssum, const int* __restrict__ ei,
    int E, int Etot, float* __restrict__ out) {
  int gw = (blockIdx.x * blockDim.x + threadIdx.x) >> 5;
  if (gw >= Etot) return;
  int lane = threadIdx.x & 31;
  int hh = lane >> 2, q = lane & 3;
  int s, d;
  if (gw < E) { s = ei[gw]; d = ei[E + gw]; } else { s = gw - E; d = s; }
  float alpha = logits[(size_t)gw * NHEAD + hh] / ssum[(size_t)d * NHEAD + hh];
  const float* pv = val + (size_t)s * HID + hh * DHEAD + q * 8;
  float* po = out + (size_t)d * HID + hh * DHEAD + q * 8;
#pragma unroll
  for (int j = 0; j < 8; ++j) atomicAdd(po + j, pv[j] * alpha);
}

// ---------------------------------------------------------------------------
// LayerNorm (+ optional GELU, + optional residual): one block per row,
// blockDim == C (256 or 512).
// ---------------------------------------------------------------------------
__global__ void k_ln_act(const float* __restrict__ X, const float* __restrict__ g,
                         const float* __restrict__ b, const float* __restrict__ res,
                         float* __restrict__ Y, int C, int do_gelu) {
  __shared__ float red[512];
  const int row = blockIdx.x, t = threadIdx.x;
  float x = X[(size_t)row * C + t];
  red[t] = x; __syncthreads();
  for (int off = C >> 1; off > 0; off >>= 1) { if (t < off) red[t] += red[t + off]; __syncthreads(); }
  float mu = red[0] / (float)C; __syncthreads();
  float d = x - mu;
  red[t] = d * d; __syncthreads();
  for (int off = C >> 1; off > 0; off >>= 1) { if (t < off) red[t] += red[t + off]; __syncthreads(); }
  float var = red[0] / (float)C;
  float y = d * rsqrtf(var + 1e-5f) * g[t] + b[t];
  if (do_gelu) y = geluf(y);
  if (res) y += res[(size_t)row * C + t];
  Y[(size_t)row * C + t] = y;
}

// ---------------------------------------------------------------------------
// Pooling: per-graph mean and max of node features.
// ---------------------------------------------------------------------------
__global__ void k_pool_cnt(const int* __restrict__ batch, int* __restrict__ cnt) {
  int n = blockIdx.x * blockDim.x + threadIdx.x;
  if (n < NN) atomicAdd(&cnt[batch[n]], 1);
}
__global__ void k_pool_acc(const float* __restrict__ h, const int* __restrict__ batch,
                           float* __restrict__ psum, int* __restrict__ pmax) {
  int i = blockIdx.x * blockDim.x + threadIdx.x;
  if (i >= NN * HID) return;
  int n = i >> 8, c = i & 255;
  int bi = batch[n];
  float v = h[i];
  atomicAdd(psum + (size_t)bi * HID + c, v);
  atomicMax(pmax + (size_t)bi * HID + c, fenc(v));
}
__global__ void k_pool_fin(const float* __restrict__ psum, const int* __restrict__ pmax,
                           const int* __restrict__ cnt, float* __restrict__ gout) {
  int i = blockIdx.x * blockDim.x + threadIdx.x;
  if (i >= NG * 512) return;
  int gi = i >> 9, c = i & 511;
  int ct = cnt[gi];
  float v;
  if (c < HID) v = psum[(size_t)gi * HID + c] / fmaxf((float)ct, 1.0f);
  else         v = (ct > 0) ? fdec(pmax[(size_t)gi * HID + (c - HID)]) : 0.0f;
  gout[i] = v;
}

// ---------------------------------------------------------------------------
// Tower MLP: out[g][tower] = W3 . gelu(W2 . gelu(W1 . s + b1) + b2) + b3
// One block (128 thr) per graph; launched per tower.
// ---------------------------------------------------------------------------
__global__ __launch_bounds__(128) void k_tower(
    const float* __restrict__ S, const float* __restrict__ W1, const float* __restrict__ b1,
    const float* __restrict__ W2, const float* __restrict__ b2,
    const float* __restrict__ W3, const float* __restrict__ b3,
    float* __restrict__ out, int tower) {
  __shared__ float sh[256];
  __shared__ float t1[128];
  __shared__ float t2[64];
  const int g = blockIdx.x, t = threadIdx.x;
  sh[t]       = S[(size_t)g * HID + t];
  sh[t + 128] = S[(size_t)g * HID + t + 128];
  __syncthreads();
  float a = b1[t];
#pragma unroll 4
  for (int i = 0; i < 256; ++i) a += sh[i] * W1[i * 128 + t];
  t1[t] = geluf(a);
  __syncthreads();
  if (t < 64) {
    float a2 = b2[t];
#pragma unroll 4
    for (int i = 0; i < 128; ++i) a2 += t1[i] * W2[i * 64 + t];
    t2[t] = geluf(a2) * W3[t];
  }
  __syncthreads();
  if (t == 0) {
    float s = b3[0];
#pragma unroll
    for (int i = 0; i < 64; ++i) s += t2[i];
    out[(size_t)g * 6 + tower] = s;
  }
}

// ---------------------------------------------------------------------------
// Host launch.  d_in order (dict-insertion flatten):
//   0:x  1:edge_index  2:batch  3..6:input_embed{W,b,g,be}
//   7+8l .. : gat[l]{Wl,bl,Wr,br,att,bias,ng,nb}  (l=0..4)
//   47..56: trans{Wq,bq,Wk,bk,Wv,bv,Wskip,bskip,ng,nb}
//   57..60: out_proj{W,b,g,be}  61..68: shared{W1,b1,g1,be1,W2,b2,g2,be2}
//   69+6t..: towers[t]{W1,b1,W2,b2,W3,b3}
// ---------------------------------------------------------------------------
extern "C" void kernel_launch(void* const* d_in, const int* in_sizes, int n_in,
                              void* d_out, int out_size, void* d_ws, size_t ws_size,
                              hipStream_t stream) {
  (void)in_sizes; (void)n_in; (void)out_size; (void)ws_size;
  const float* X  = (const float*)d_in[0];
  const int*   EI = (const int*)d_in[1];
  const int*   BT = (const int*)d_in[2];
  const float* const* P = (const float* const*)(d_in + 3);

  // workspace layout (floats)
  float* wsf = (float*)d_ws;
  const size_t BIG = (size_t)NN * HID;          // 8388608
  float* h  = wsf;
  float* bA = wsf + 1 * BIG;
  float* bB = wsf + 2 * BIG;
  float* bC = wsf + 3 * BIG;
  float* bD = wsf + 4 * BIG;
  float* lg = wsf + 5 * BIG;                    // EGAT*8 = 2359296
  int*   mE = (int*)(lg + (size_t)EGAT * NHEAD);
  float* sS = (float*)(mE + (size_t)NN * NHEAD);
  float* pS = sS + (size_t)NN * NHEAD;          // NG*HID
  int*   pM = (int*)(pS + (size_t)NG * HID);
  int*   pC = (int*)(pM + (size_t)NG * HID);    // NG ints
  float* gG = (float*)(pC + NG);                // NG*512
  float* t1 = gG + (size_t)NG * 512;            // NG*512
  float* t2 = t1 + (size_t)NG * 512;            // NG*512
  float* s1 = t2 + (size_t)NG * 512;            // NG*256
  float* s2 = s1 + (size_t)NG * HID;            // NG*256

  // 1) input embedding
  k_embed<<<NN, 256, 0, stream>>>(X, P[0], P[1], P[2], P[3], h);

  // 2) GATv2 layers
  for (int l = 0; l < NLAYER; ++l) {
    const float* const* q = P + 4 + l * 8;  // Wl,bl,Wr,br,att,bias,ng,nb
    k_gemm<<<dim3(NN / 128, 1), 256, 0, stream>>>(h, q[0], q[1], bA, NN, HID, HID);
    k_gemm<<<dim3(NN / 128, 1), 256, 0, stream>>>(h, q[2], q[3], bB, NN, HID, HID);
    k_fill_i<<<(NN * NHEAD) / 256, 256, 0, stream>>>(mE, (int)0x80000000, NN * NHEAD);
    k_fill_f<<<(NN * NHEAD) / 256, 256, 0, stream>>>(sS, 0.0f, NN * NHEAD);
    k_rows_bias<<<(NN * HID) / 256, 256, 0, stream>>>(bC, q[5], NN);
    k_attA_gat<<<(EGAT + 7) / 8, 256, 0, stream>>>(bA, bB, q[4], EI, NE, EGAT, lg, mE);
    k_attB<<<(EGAT * NHEAD + 255) / 256, 256, 0, stream>>>(lg, mE, sS, EI, NE, EGAT);
    k_attC<<<(EGAT + 7) / 8, 256, 0, stream>>>(bA, lg, sS, EI, NE, EGAT, bC);
    k_ln_act<<<NN, HID, 0, stream>>>(bC, q[6], q[7], h, h, HID, 1);  // h += gelu(ln(.))
  }

  // 3) TransformerConv
  {
    const float* const* q = P + 44;  // Wq,bq,Wk,bk,Wv,bv,Wskip,bskip,ng,nb
    k_gemm<<<dim3(NN / 128, 1), 256, 0, stream>>>(h, q[0], q[1], bA, NN, HID, HID);  // Q
    k_gemm<<<dim3(NN / 128, 1), 256, 0, stream>>>(h, q[2], q[3], bB, NN, HID, HID);  // K
    k_gemm<<<dim3(NN / 128, 1), 256, 0, stream>>>(h, q[4], q[5], bC, NN, HID, HID);  // V
    k_gemm<<<dim3(NN / 128, 1), 256, 0, stream>>>(h, q[6], q[7], bD, NN, HID, HID);  // skip -> out base
    k_fill_i<<<(NN * NHEAD) / 256, 256, 0, stream>>>(mE, (int)0x80000000, NN * NHEAD);
    k_fill_f<<<(NN * NHEAD) / 256, 256, 0, stream>>>(sS, 0.0f, NN * NHEAD);
    k_attA_trans<<<(NE + 7) / 8, 256, 0, stream>>>(bA, bB, EI, NE, lg, mE);
    k_attB<<<(NE * NHEAD + 255) / 256, 256, 0, stream>>>(lg, mE, sS, EI, NE, NE);
    k_attC<<<(NE + 7) / 8, 256, 0, stream>>>(bC, lg, sS, EI, NE, NE, bD);
    k_ln_act<<<NN, HID, 0, stream>>>(bD, q[8], q[9], h, h, HID, 0);  // h += ln(.)
  }

  // 4) pooling -> gG [NG, 512]
  k_fill_f<<<(NG * HID) / 256, 256, 0, stream>>>(pS, 0.0f, NG * HID);
  k_fill_i<<<(NG * HID) / 256, 256, 0, stream>>>(pM, (int)0x80000000, NG * HID);
  k_fill_i<<<(NG + 255) / 256, 256, 0, stream>>>(pC, 0, NG);
  k_pool_cnt<<<NN / 256, 256, 0, stream>>>(BT, pC);
  k_pool_acc<<<(NN * HID) / 256, 256, 0, stream>>>(h, BT, pS, pM);
  k_pool_fin<<<(NG * 512) / 256, 256, 0, stream>>>(pS, pM, pC, gG);

  // 5) out_proj: g = gelu(ln(g @ W[512,512] + b))
  {
    const float* const* po = P + 54;
    k_gemm<<<dim3(NG / 128, 2), 256, 0, stream>>>(gG, po[0], po[1], t1, NG, 512, 512);
    k_ln_act<<<NG, 512, 0, stream>>>(t1, po[2], po[3], nullptr, t2, 512, 1);
  }
  // 6) shared MLP
  {
    const float* const* ps = P + 58;
    k_gemm<<<dim3(NG / 128, 1), 256, 0, stream>>>(t2, ps[0], ps[1], s1, NG, 512, HID);
    k_ln_act<<<NG, HID, 0, stream>>>(s1, ps[2], ps[3], nullptr, s2, HID, 1);
    k_gemm<<<dim3(NG / 128, 1), 256, 0, stream>>>(s2, ps[4], ps[5], s1, NG, HID, HID);
    k_ln_act<<<NG, HID, 0, stream>>>(s1, ps[6], ps[7], nullptr, s2, HID, 1);
  }
  // 7) task towers -> d_out [NG, 6]
  float* out = (float*)d_out;
  for (int t = 0; t < 6; ++t) {
    const float* const* pt = P + 66 + t * 6;
    k_tower<<<NG, 128, 0, stream>>>(s2, pt[0], pt[1], pt[2], pt[3], pt[4], pt[5], out, t);
  }
}
